// MultiHeadedAttention_14388140441659
// MI455X (gfx1250) — matrix operand order repaired and checked by reading, not compiled, not run on hardware
//
#include <hip/hip_runtime.h>
#include <hip/hip_bf16.h>

// ---- problem constants (match reference) ----
#define BB 4
#define TT 2048
#define EE 1024
#define HH 16
#define DD 64

typedef __attribute__((ext_vector_type(16))) __bf16        v16bf;
typedef __attribute__((ext_vector_type(2)))  __bf16        v2bf;
typedef __attribute__((ext_vector_type(8)))  float         v8f;
typedef __attribute__((ext_vector_type(8)))  unsigned int  v8u;

// ---- f32 -> bf16 via native fptrunc (RTNE) ----
__device__ __forceinline__ unsigned short f2bf(float f) {
  __bf16 h = (__bf16)f;
  return __builtin_bit_cast(unsigned short, h);
}
__device__ __forceinline__ unsigned int pk2bf(float lo, float hi) {
  v2bf r;
  r[0] = (__bf16)lo;
  r[1] = (__bf16)hi;
  return __builtin_bit_cast(unsigned int, r);
}

// fragment = two contiguous 16-byte spans
__device__ __forceinline__ v16bf frag_u4(uint4 a, uint4 b) {
  v8u t;
  t[0] = a.x; t[1] = a.y; t[2] = a.z; t[3] = a.w;
  t[4] = b.x; t[5] = b.y; t[6] = b.z; t[7] = b.w;
  return __builtin_bit_cast(v16bf, t);
}
__device__ __forceinline__ uint4 ld4(const unsigned short* p) {
  return *(const uint4*)p;
}

__device__ __forceinline__ v8f wmma_bf16(v16bf a, v16bf b, v8f c) {
  return __builtin_amdgcn_wmma_f32_16x16x32_bf16(false, a, false, b, (short)0, c,
                                                 false, false);
}
__device__ __forceinline__ void lds_fence_wave() {
  __builtin_amdgcn_wave_barrier();
  asm volatile("s_wait_dscnt 0" ::: "memory");
  __builtin_amdgcn_wave_barrier();
}

// =====================================================================
// Kernel 0a: straight f32 -> bf16 convert (8 elements / thread)
// =====================================================================
__global__ __launch_bounds__(256) void cvt_kernel(const float* __restrict__ src,
                                                  unsigned short* __restrict__ dst,
                                                  int n8) {
  int i = blockIdx.x * 256 + threadIdx.x;
  if (i < n8) {
    const float4* s = (const float4*)src + 2 * (size_t)i;
    float4 a = s[0], b = s[1];
    uint4 o;
    o.x = pk2bf(a.x, a.y); o.y = pk2bf(a.z, a.w);
    o.z = pk2bf(b.x, b.y); o.w = pk2bf(b.z, b.w);
    ((uint4*)dst)[i] = o;
  }
}

// =====================================================================
// Kernel 0b: transpose-convert Wq/Wk/Wv [h][e][d] f32 -> WT [pj*H+h][d][e] bf16
// grid = (E/64, 3*H), block 256; 64e x 64d tile through LDS.
// =====================================================================
__global__ __launch_bounds__(256) void wtrans_kernel(
    const float* __restrict__ Wq, const float* __restrict__ Wk,
    const float* __restrict__ Wv, unsigned short* __restrict__ wt) {
  __shared__ __align__(16) unsigned short Tt[DD][72];  // [d][e], padded
  const int e0 = blockIdx.x * 64;
  const int zz = blockIdx.y;            // 0..3*H-1
  const int pj = zz / HH, h = zz % HH;
  const float* W = (pj == 0 ? Wq : (pj == 1 ? Wk : Wv)) + (size_t)h * EE * DD;
#pragma unroll
  for (int it = 0; it < 16; ++it) {
    int i = threadIdx.x + it * 256;     // 0..4095 : e=i>>6, d=i&63 (coalesced in d)
    int e = i >> 6, d = i & 63;
    Tt[d][e] = f2bf(W[(size_t)(e0 + e) * DD + d]);
  }
  __syncthreads();
  unsigned short* outp = wt + (size_t)zz * DD * EE;
#pragma unroll
  for (int it = 0; it < 8; ++it) {
    int i = threadIdx.x + it * 256;     // 0..2047 : d=i>>5, e-pair=i&31
    int d = i >> 5, pe = i & 31;
    unsigned int u = *(const unsigned int*)&Tt[d][2 * pe];
    *(unsigned int*)&outp[(size_t)d * EE + e0 + 2 * pe] = u;
  }
}

// =====================================================================
// Kernel 1: q/k/v projection, pure b128->wmma loop (no LDS, no packing).
// grid = (T/128, B, 3*H), block = 256.
//   pj=0 -> q[B,H,T,D] ; pj=1 -> k[B,H,T,D] ; pj=2 -> vT[B,H,D,T]  (bf16)
// =====================================================================
__global__ __launch_bounds__(256) void qkv_kernel(
    const unsigned short* __restrict__ xbf, const unsigned short* __restrict__ wt,
    unsigned short* __restrict__ qw, unsigned short* __restrict__ kw,
    unsigned short* __restrict__ vtw) {
  const int b  = blockIdx.y;
  const int zz = blockIdx.z;
  const int pj = zz / HH;
  const int h  = zz % HH;
  const int w  = threadIdx.x >> 5;
  const int l  = threadIdx.x & 31;
  const int g  = l >> 4;
  const int ln = l & 15;
  const int row0 = blockIdx.x * 128 + w * 16;

  const unsigned short* W = wt + (size_t)zz * DD * EE;        // [d][e] bf16
  const size_t xrow = ((size_t)b * TT + row0 + ln) * EE;

  v8f acc[4] = {v8f{}, v8f{}, v8f{}, v8f{}};

  for (int kc = 0; kc < EE; kc += 32) {
    v16bf af = frag_u4(ld4(&xbf[xrow + kc + g * 16]),
                       ld4(&xbf[xrow + kc + g * 16 + 8]));
#pragma unroll
    for (int nt = 0; nt < 4; ++nt) {
      int d = nt * 16 + ln;
      const unsigned short* wr = W + (size_t)d * EE + kc + g * 16;
      acc[nt] = wmma_bf16(af, frag_u4(ld4(wr), ld4(wr + 8)), acc[nt]);
    }
  }

  // ---- branchless store (C-layout: row = r+8g, col = ln) ----
  unsigned short* out = (pj == 0) ? qw : (pj == 1 ? kw : vtw);
  const size_t obase = ((size_t)b * HH + h) * TT * DD;
  const size_t sT = (pj < 2) ? (size_t)DD : (size_t)1;
  const size_t sD = (pj < 2) ? (size_t)1 : (size_t)TT;
#pragma unroll
  for (int nt = 0; nt < 4; ++nt) {
#pragma unroll
    for (int r = 0; r < 8; ++r) {
      size_t t = row0 + r + 8 * g;
      size_t d = nt * 16 + ln;
      out[obase + t * sT + d * sD] = f2bf(acc[nt][r]);
    }
  }
}

// =====================================================================
// Kernel 2: causal flash attention per (b,h).  grid = (T/128, B*H), block 256.
// =====================================================================
__global__ __launch_bounds__(256) void attn_kernel(
    const unsigned short* __restrict__ qw, const unsigned short* __restrict__ kw,
    const unsigned short* __restrict__ vtw, unsigned short* __restrict__ zw) {
  __shared__ __align__(16) unsigned short P[8][16][32];  // per-wave 16x32 bf16

  const int w  = threadIdx.x >> 5;
  const int l  = threadIdx.x & 31;
  const int g  = l >> 4;
  const int ln = l & 15;
  const int bh = blockIdx.y;
  const int b  = bh / HH;
  const int h  = bh % HH;
  const int row0 = blockIdx.x * 128 + w * 16;

  const unsigned short* qb = qw + (size_t)bh * TT * DD;
  const unsigned short* kb = kw + (size_t)bh * TT * DD;
  const unsigned short* vb = vtw + (size_t)bh * DD * TT;

  // Q fragments (A-layout), resident: D=64 -> 2 frags, 2x b128 each
  const size_t qrow = (size_t)(row0 + ln) * DD;
  const v16bf qf0 = frag_u4(ld4(&qb[qrow + g * 16]), ld4(&qb[qrow + g * 16 + 8]));
  const v16bf qf1 = frag_u4(ld4(&qb[qrow + 32 + g * 16]), ld4(&qb[qrow + 32 + g * 16 + 8]));

  v8f o[4] = {v8f{}, v8f{}, v8f{}, v8f{}};
  float mrun[8], lrun[8];
#pragma unroll
  for (int r = 0; r < 8; ++r) { mrun[r] = -1e30f; lrun[r] = 0.0f; }

  const float scale = 0.125f;          // 1/sqrt(D)
  const int kmax = row0 + 16;          // wave-uniform causal bound

  for (int k0 = 0; k0 < kmax; k0 += 32) {
    // ---- scores for two 16-key subtiles ----
    v8f s[2];
#pragma unroll
    for (int sub = 0; sub < 2; ++sub) {
      const int kcol = k0 + sub * 16 + ln;
      const size_t kr = (size_t)kcol * DD;
      v16bf kf0 = frag_u4(ld4(&kb[kr + g * 16]), ld4(&kb[kr + g * 16 + 8]));
      v16bf kf1 = frag_u4(ld4(&kb[kr + 32 + g * 16]), ld4(&kb[kr + 32 + g * 16 + 8]));
      v8f sa = v8f{};
      sa = wmma_bf16(qf0, kf0, sa);
      sa = wmma_bf16(qf1, kf1, sa);
#pragma unroll
      for (int r = 0; r < 8; ++r) {
        int qi = row0 + r + 8 * g;
        s[sub][r] = (kcol <= qi) ? sa[r] * scale : -1e30f;
      }
    }

    // ---- online softmax per row (rows live across a 16-lane half) ----
#pragma unroll
    for (int r = 0; r < 8; ++r) {
      float mt = fmaxf(s[0][r], s[1][r]);
#pragma unroll
      for (int xm = 1; xm < 16; xm <<= 1) mt = fmaxf(mt, __shfl_xor(mt, xm, 32));
      float mnew = fmaxf(mrun[r], mt);
      float p0 = __expf(s[0][r] - mnew);
      float p1 = __expf(s[1][r] - mnew);
      float rs = p0 + p1;
#pragma unroll
      for (int xm = 1; xm < 16; xm <<= 1) rs += __shfl_xor(rs, xm, 32);
      float corr = __expf(mrun[r] - mnew);
      lrun[r] = lrun[r] * corr + rs;
      mrun[r] = mnew;
#pragma unroll
      for (int nt = 0; nt < 4; ++nt) o[nt][r] *= corr;
      P[w][r + 8 * g][ln]      = f2bf(p0);
      P[w][r + 8 * g][16 + ln] = f2bf(p1);
    }
    lds_fence_wave();

    // ---- re-read P in A-layout: 2x ds b128 ----
    const unsigned short* pw = &P[w][0][0];
    v16bf pf = frag_u4(ld4(&pw[ln * 32 + g * 16]), ld4(&pw[ln * 32 + g * 16 + 8]));
    lds_fence_wave();

    // ---- o += P(16x32) @ V(32x64): vT pairs contiguous, 2x b128 per frag ----
#pragma unroll
    for (int nt = 0; nt < 4; ++nt) {
      int d = nt * 16 + ln;
      const size_t vr = (size_t)d * TT + k0 + g * 16;
      v16bf vf = frag_u4(ld4(&vb[vr]), ld4(&vb[vr + 8]));
      o[nt] = wmma_bf16(pf, vf, o[nt]);
    }
  }

  // ---- normalize + write z[B,T,E] (head-concat) as bf16 ----
  float inv[8];
#pragma unroll
  for (int r = 0; r < 8; ++r) inv[r] = 1.0f / lrun[r];
#pragma unroll
  for (int nt = 0; nt < 4; ++nt) {
#pragma unroll
    for (int r = 0; r < 8; ++r) {
      int t = row0 + r + 8 * g;
      int d = nt * 16 + ln;
      zw[((size_t)b * TT + t) * EE + h * DD + d] = f2bf(o[nt][r] * inv[r]);
    }
  }
}

// =====================================================================
// Kernel 3: out = z @ Wp^T + bp.  grid = (B*T/128, E/64), block 256.
// Wp pre-converted to bf16 (rows K-contiguous) -> pure b128->wmma loop.
// =====================================================================
__global__ __launch_bounds__(256) void proj_kernel(
    const unsigned short* __restrict__ zw, const unsigned short* __restrict__ wpbf,
    const float* __restrict__ bp, float* __restrict__ out) {
  const int w  = threadIdx.x >> 5;
  const int l  = threadIdx.x & 31;
  const int g  = l >> 4;
  const int ln = l & 15;
  const int row0 = blockIdx.x * 128 + w * 16;   // flat b*T index
  const int e0   = blockIdx.y * 64;

  v8f acc[4];
#pragma unroll
  for (int nt = 0; nt < 4; ++nt) {
    float bias = bp[e0 + nt * 16 + ln];
#pragma unroll
    for (int r = 0; r < 8; ++r) acc[nt][r] = bias;
  }

  const size_t zrow = (size_t)(row0 + ln) * EE;
  for (int kc = 0; kc < EE; kc += 32) {
    v16bf af = frag_u4(ld4(&zw[zrow + kc + g * 16]),
                       ld4(&zw[zrow + kc + g * 16 + 8]));
#pragma unroll
    for (int nt = 0; nt < 4; ++nt) {
      int e = e0 + nt * 16 + ln;
      const unsigned short* wr = wpbf + (size_t)e * EE + kc + g * 16;
      __builtin_prefetch(wr + 64, 0, 0);        // global_prefetch path
      acc[nt] = wmma_bf16(af, frag_u4(ld4(wr), ld4(wr + 8)), acc[nt]);
    }
  }

#pragma unroll
  for (int nt = 0; nt < 4; ++nt) {
#pragma unroll
    for (int r = 0; r < 8; ++r) {
      out[(size_t)(row0 + r + 8 * g) * EE + e0 + nt * 16 + ln] = acc[nt][r];
    }
  }
}

extern "C" void kernel_launch(void* const* d_in, const int* in_sizes, int n_in,
                              void* d_out, int out_size, void* d_ws, size_t ws_size,
                              hipStream_t stream) {
  (void)in_sizes; (void)n_in; (void)out_size; (void)ws_size;
  const float* x  = (const float*)d_in[0];
  const float* Wq = (const float*)d_in[1];
  const float* Wk = (const float*)d_in[2];
  const float* Wv = (const float*)d_in[3];
  const float* Wp = (const float*)d_in[4];
  const float* bp = (const float*)d_in[5];
  float* out = (float*)d_out;

  // workspace layout (bf16 halves):
  //   q | k | vT | z : 4 x 8M elems   then xbf 8M, WT 3M, Wpbf 1M  (~92 MB)
  unsigned short* ws = (unsigned short*)d_ws;
  const size_t qN = (size_t)BB * HH * TT * DD;        // 8,388,608
  unsigned short* qw   = ws;
  unsigned short* kw   = ws + qN;
  unsigned short* vtw  = ws + 2 * qN;
  unsigned short* zw   = ws + 3 * qN;
  unsigned short* xbf  = ws + 4 * qN;                 // B*T*E elems (== qN)
  unsigned short* wtbf = ws + 5 * qN;                 // 3*H*D*E = 3,145,728
  unsigned short* wpbf = wtbf + (size_t)3 * HH * DD * EE;  // E*E = 1,048,576

  dim3 blk(256);
  // one-time conversions
  {
    int n8x = (int)((size_t)BB * TT * EE / 8);        // 1,048,576
    cvt_kernel<<<dim3((n8x + 255) / 256), blk, 0, stream>>>(x, xbf, n8x);
    int n8p = (int)((size_t)EE * EE / 8);             // 131,072
    cvt_kernel<<<dim3((n8p + 255) / 256), blk, 0, stream>>>(Wp, wpbf, n8p);
    wtrans_kernel<<<dim3(EE / 64, 3 * HH), blk, 0, stream>>>(Wq, Wk, Wv, wtbf);
  }
  // main pipeline
  qkv_kernel<<<dim3(TT / 128, BB, 3 * HH), blk, 0, stream>>>(xbf, wtbf, qw, kw, vtw);
  attn_kernel<<<dim3(TT / 128, BB * HH), blk, 0, stream>>>(qw, kw, vtw, zw);
  proj_kernel<<<dim3((BB * TT) / 128, EE / 64), blk, 0, stream>>>(zw, wpbf, bp, out);
}
